// MultiHeadDifferentialAttention_27066883899491
// MI455X (gfx1250) — matrix operand order repaired and checked by reading, not compiled, not run on hardware
//
#include <hip/hip_runtime.h>
#include <hip/hip_bf16.h>

#define NH_ 16
#define S_  2048
#define HD_ 128

typedef __attribute__((ext_vector_type(16))) __bf16 bf16x16;
typedef __attribute__((ext_vector_type(8)))  __bf16 bf16x8;
typedef __attribute__((ext_vector_type(8)))  float  v8f;
typedef __attribute__((ext_vector_type(4)))  unsigned u32x4;
typedef __attribute__((ext_vector_type(8)))  int  i32x8;
typedef __attribute__((ext_vector_type(4)))  int  i32x4;

// ---------------------------------------------------------------------------
// WMMA helpers (CDNA5 V_WMMA_F32_16X16X32_BF16, wave32)
// ---------------------------------------------------------------------------
__device__ __forceinline__ v8f wmma_bf16(bf16x16 a, bf16x16 b, v8f c) {
  return __builtin_amdgcn_wmma_f32_16x16x32_bf16(false, a, false, b,
                                                 (short)0, c, false, false);
}

// A-matrix 16x32 bf16 fragment.  lane = (m = lane&15, half = lane>>4)
__device__ __forceinline__ bf16x16 frag_a(const __bf16* row, int k, int half) {
  bf16x8 lo = *(const bf16x8*)(row + k + half * 8);
  bf16x8 hi = *(const bf16x8*)(row + k + 16 + half * 8);
  bf16x16 r;
#pragma unroll
  for (int i = 0; i < 8; ++i) { r[i] = lo[i]; r[8 + i] = hi[i]; }
  return r;
}

// B-matrix 32x16 bf16 fragment from an (N,K) row-major operand.
__device__ __forceinline__ bf16x16 frag_b(const __bf16* row, int k, int half) {
  bf16x8 lo = *(const bf16x8*)(row + k + half * 8);
  bf16x8 hi = *(const bf16x8*)(row + k + 16 + half * 8);
  bf16x16 r;
#pragma unroll
  for (int i = 0; i < 8; ++i) { r[i] = lo[i]; r[8 + i] = hi[i]; }
  return r;
}

// CDNA5 async global -> LDS, 16B per lane, tracked by ASYNCcnt.
__device__ __forceinline__ void async_copy16(unsigned lds_addr,
                                             const __bf16* gaddr) {
  asm volatile("global_load_async_to_lds_b128 %0, %1, off"
               :
               : "v"(lds_addr), "v"(gaddr)
               : "memory");
}

// CDNA5 Tensor Data Mover: 2D tile (tile_h rows x tile_w elems, bf16,
// row stride `row_stride` elems) from global -> packed LDS rows.
// One descriptor, one instruction per wave; tracked by TENSORcnt.
__device__ __forceinline__ void tdm_load_2d(const __bf16* gaddr,
                                            unsigned lds_addr, unsigned tile_w,
                                            unsigned tile_h,
                                            unsigned row_stride) {
  const unsigned long long ga = (unsigned long long)(size_t)gaddr;
  u32x4 g0;
  g0[0] = 1u;                                   // count=1 (valid user D#)
  g0[1] = lds_addr;                             // lds_addr (bytes)
  g0[2] = (unsigned)(ga & 0xFFFFFFFFu);         // global_addr[31:0]
  g0[3] = (unsigned)((ga >> 32) & 0x01FFFFFFu)  // global_addr[56:32]
          | 0x80000000u;                        // type = 2 ("image")
  i32x8 g1;
  g1[0] = (int)(1u << 16);                      // data_size=1 (2B), mask=0
  g1[1] = (int)((row_stride & 0xFFFFu) << 16);  // tensor_dim0[15:0]
  g1[2] = (int)(((row_stride >> 16) & 0xFFFFu)  // tensor_dim0[31:16]
                | ((tile_h & 0xFFFFu) << 16));  // tensor_dim1[15:0]
  g1[3] = (int)((tile_w & 0xFFFFu) << 16);      // tensor_dim1 hi=0 | tile_dim0
  g1[4] = (int)(tile_h & 0xFFFFu);              // tile_dim1 (tile_dim2=0)
  g1[5] = (int)row_stride;                      // tensor_dim0_stride[31:0]
  g1[6] = 0;                                    // stride hi | dim1_stride lo
  g1[7] = 0;
  i32x4 z4;
  z4[0] = z4[1] = z4[2] = z4[3] = 0;            // 2D: groups 2/3 unused
  i32x8 z8;
#pragma unroll
  for (int i = 0; i < 8; ++i) z8[i] = 0;
  // 6-arg form (clang-23 / therock-10.0 toolchain)
  __builtin_amdgcn_tensor_load_to_lds(g0, g1, z4, z4, z8, 0);
}

// ---------------------------------------------------------------------------
// f32 -> bf16 conversion
// ---------------------------------------------------------------------------
__global__ void cvt_bf16_kernel(const float* __restrict__ in,
                                __bf16* __restrict__ out, size_t n) {
  size_t i = (size_t)blockIdx.x * blockDim.x + threadIdx.x;
  if (i < n) out[i] = (__bf16)in[i];
}

// ---------------------------------------------------------------------------
// C = A (M,K) * W(N,K)^T ; bf16 in, f32 or bf16 out.
// Block = 256 threads = 8 waves (4m x 2n); block tile 128 x 128; K-step 32.
// A tile:  TDM tensor_load_to_lds (wave 0), TENSORcnt double buffered.
// B tile:  global_load_async_to_lds_b128 (all lanes), ASYNCcnt double buffered.
// ---------------------------------------------------------------------------
template <bool F32OUT>
__global__ __launch_bounds__(256) void gemm_bf16_kernel(
    const __bf16* __restrict__ A, const __bf16* __restrict__ W,
    void* __restrict__ Cout, int M, int N, int K) {
  __shared__ alignas(16) __bf16 Abuf[2][128 * 32];
  __shared__ alignas(16) __bf16 Bbuf[2][128 * 32];

  const int lane = threadIdx.x & 31;
  const int wave = threadIdx.x >> 5;
  const int ml = lane & 15, half = lane >> 4;
  const int m0 = blockIdx.x * 128;
  const int n0 = blockIdx.y * 128;
  const int wm0 = (wave >> 1) * 32;  // wave m-offset inside block tile
  const int wn0 = (wave & 1) * 64;   // wave n-offset inside block tile

  // per-thread async-copy mapping for the B tile: rows r0 and r0+64,
  // one 16B chunk each -> 2 async ops / thread / K-step
  const int tid = threadIdx.x;
  const int r0 = tid >> 2;
  const int c0 = (tid & 3) * 8;

  auto issue_a_tdm = [&](int buf, int kk) {
    tdm_load_2d(A + (size_t)m0 * K + kk,
                (unsigned)(size_t)&Abuf[buf][0], 32u, 128u, (unsigned)K);
  };
  auto issue_b_async = [&](int buf, int kk) {
    const __bf16* gb = W + (size_t)(n0 + r0) * K + kk + c0;
    async_copy16((unsigned)(size_t)&Bbuf[buf][r0 * 32 + c0], gb);
    async_copy16((unsigned)(size_t)&Bbuf[buf][(r0 + 64) * 32 + c0],
                 gb + (size_t)64 * K);
  };

  v8f zero8;
#pragma unroll
  for (int i = 0; i < 8; ++i) zero8[i] = 0.0f;

  v8f acc[2][4];
#pragma unroll
  for (int i = 0; i < 2; ++i)
#pragma unroll
    for (int j = 0; j < 4; ++j) acc[i][j] = zero8;

  if (wave == 0) issue_a_tdm(0, 0);
  issue_b_async(0, 0);

  const int nK = K / 32;
  for (int kt = 0; kt < nK; ++kt) {
    const int cur = kt & 1;
    if (kt + 1 < nK) {
      if (wave == 0) issue_a_tdm(cur ^ 1, (kt + 1) * 32);
      issue_b_async(cur ^ 1, (kt + 1) * 32);
      if (wave == 0) __builtin_amdgcn_s_wait_tensorcnt((short)1);
      asm volatile("s_wait_asynccnt 2" ::: "memory");
    } else {
      if (wave == 0) __builtin_amdgcn_s_wait_tensorcnt((short)0);
      asm volatile("s_wait_asynccnt 0" ::: "memory");
    }
    __syncthreads();

    bf16x16 af[2], bfm[4];
#pragma unroll
    for (int mt = 0; mt < 2; ++mt)
      af[mt] = frag_a(&Abuf[cur][(wm0 + mt * 16 + ml) * 32], 0, half);
#pragma unroll
    for (int nt = 0; nt < 4; ++nt)
      bfm[nt] = frag_b(&Bbuf[cur][(wn0 + nt * 16 + ml) * 32], 0, half);
#pragma unroll
    for (int mt = 0; mt < 2; ++mt)
#pragma unroll
      for (int nt = 0; nt < 4; ++nt)
        acc[mt][nt] = wmma_bf16(af[mt], bfm[nt], acc[mt][nt]);

    __syncthreads();  // protect buffer reuse next iteration
  }

#pragma unroll
  for (int mt = 0; mt < 2; ++mt)
#pragma unroll
    for (int nt = 0; nt < 4; ++nt)
#pragma unroll
      for (int v = 0; v < 8; ++v) {
        const int row = m0 + wm0 + mt * 16 + v + 8 * half;
        const int col = n0 + wn0 + nt * 16 + ml;
        if constexpr (F32OUT)
          ((float*)Cout)[(size_t)row * N + col] = acc[mt][nt][v];
        else
          ((__bf16*)Cout)[(size_t)row * N + col] = (__bf16)acc[mt][nt][v];
      }
}

// ---------------------------------------------------------------------------
// RoPE + reorg: q,k,v bf16 (b,s,h,256) -> q1r/q2r/k1r/k2r bf16 (b,h,S,128)
// and vT bf16 (b,h,256,S).  One block per (b,s,h).
// ---------------------------------------------------------------------------
__global__ __launch_bounds__(256) void rope_reorg_kernel(
    const __bf16* __restrict__ q, const __bf16* __restrict__ k,
    const __bf16* __restrict__ v, const float* __restrict__ fcos,
    const float* __restrict__ fsin, __bf16* __restrict__ q1r,
    __bf16* __restrict__ q2r, __bf16* __restrict__ k1r,
    __bf16* __restrict__ k2r, __bf16* __restrict__ vT) {
  const int d = threadIdx.x;
  const size_t bsh = blockIdx.x;  // ((b*S)+s)*NH + h
  const int h = (int)(bsh % NH_);
  const size_t bs = bsh / NH_;
  const int s = (int)(bs % S_);
  const int b = (int)(bs / S_);
  const size_t in_off = bsh * 256 + d;
  const int bhh = b * NH_ + h;

  vT[((size_t)bhh * 256 + d) * S_ + s] = v[in_off];

  const int dd = d & 127;
  float c = 0.f, sn = 0.f;
  if (dd < 64) {
    const int j = dd >> 1;
    c = fcos[s * 32 + j];
    sn = fsin[s * 32 + j];
  }

#pragma unroll
  for (int which = 0; which < 2; ++which) {
    const __bf16* src = which ? k : q;
    float out;
    if (dd < 64) {
      const size_t base = bsh * 256 + (size_t)(d & ~1);
      const float xr = (float)src[base];
      const float xi = (float)src[base + 1];
      out = (d & 1) ? (xr * sn + xi * c) : (xr * c - xi * sn);
    } else {
      out = (float)src[in_off];
    }
    __bf16* dst;
    if (which == 0) dst = (d < 128) ? q1r : q2r;
    else            dst = (d < 128) ? k1r : k2r;
    dst[((size_t)bhh * S_ + s) * HD_ + dd] = (__bf16)out;
  }
}

// ---------------------------------------------------------------------------
// lam[h] = exp(sum lq1*lk1) - exp(sum lq2*lk2) + lambda_init
// ---------------------------------------------------------------------------
__global__ void lambda_kernel(const float* __restrict__ lq1,
                              const float* __restrict__ lk1,
                              const float* __restrict__ lq2,
                              const float* __restrict__ lk2,
                              float* __restrict__ lam) {
  const int h = threadIdx.x;
  if (h < NH_) {
    float s1 = 0.f, s2 = 0.f;
    for (int d = 0; d < HD_; ++d) {
      s1 += lq1[h * HD_ + d] * lk1[h * HD_ + d];
      s2 += lq2[h * HD_ + d] * lk2[h * HD_ + d];
    }
    const float lam0 = 0.8f - 0.6f * __expf(-3.6f);
    lam[h] = __expf(s1) - __expf(s2) + lam0;
  }
}

// ---------------------------------------------------------------------------
// Causal flash attention for one branch.
// Q,K: (b*h, S, 128) bf16.  Vt: (b*h, 256, S) bf16.  O: (b*h, S, 256) bf16.
// 8 waves/block, each wave owns 16 queries; streaming softmax; P tile staged
// through LDS (C-layout -> A-layout).  Mask only applied on diagonal tiles.
// ---------------------------------------------------------------------------
__global__ __launch_bounds__(256) void flash_attn_kernel(
    const __bf16* __restrict__ Q, const __bf16* __restrict__ Km,
    const __bf16* __restrict__ Vt, __bf16* __restrict__ O) {
  __shared__ alignas(16) __bf16 pshare[8][16][32];
  const int lane = threadIdx.x & 31;
  const int wave = threadIdx.x >> 5;
  const int ml = lane & 15, half = lane >> 4;
  const int bh = blockIdx.y;
  const __bf16* Qh = Q + (size_t)bh * S_ * HD_;
  const __bf16* Kh = Km + (size_t)bh * S_ * HD_;
  const __bf16* Vh = Vt + (size_t)bh * 256 * S_;
  const int q0 = blockIdx.x * 128 + wave * 16;

  v8f zero8;
#pragma unroll
  for (int i = 0; i < 8; ++i) zero8[i] = 0.0f;

  bf16x16 aq[4];
#pragma unroll
  for (int ks = 0; ks < 4; ++ks)
    aq[ks] = frag_a(Qh + (size_t)(q0 + ml) * HD_, ks * 32, half);

  v8f acc[16];
#pragma unroll
  for (int t = 0; t < 16; ++t) acc[t] = zero8;
  float mst[8], lst[8];
#pragma unroll
  for (int v = 0; v < 8; ++v) { mst[v] = -1e30f; lst[v] = 0.0f; }

  const float scale = 0.08838834764831845f;  // 128^-0.5
  const int ktmax = (q0 + 15) >> 5;
  for (int kt = 0; kt <= ktmax; ++kt) {
    const int kb = kt * 32;
    if (kt < ktmax) {  // prefetch next K / V tiles (global_prefetch_b8)
      __builtin_prefetch(Kh + (size_t)(kb + 32 + ml) * HD_, 0, 1);
      __builtin_prefetch(Vh + (size_t)(ml * 16) * S_ + kb + 32, 0, 1);
    }
    // ---- scores: 16 queries x 32 keys over d=128 ----
    v8f sc0 = zero8, sc1 = zero8;
#pragma unroll
    for (int ks = 0; ks < 4; ++ks) {
      bf16x16 b0 = frag_b(Kh + (size_t)(kb + ml) * HD_, ks * 32, half);
      bf16x16 b1 = frag_b(Kh + (size_t)(kb + 16 + ml) * HD_, ks * 32, half);
      sc0 = wmma_bf16(aq[ks], b0, sc0);
      sc1 = wmma_bf16(aq[ks], b1, sc1);
    }
    // ---- scale (+ causal mask only on diagonal tiles) + online softmax ----
    const bool need_mask = (kb + 31 > q0);  // wave-uniform
#pragma unroll
    for (int v = 0; v < 8; ++v) {
      const int qrow = q0 + v + 8 * half;
      float s0 = sc0[v] * scale;
      float s1 = sc1[v] * scale;
      if (need_mask) {
        if (kb + ml > qrow) s0 = -1e30f;
        if (kb + 16 + ml > qrow) s1 = -1e30f;
      }
      float rm = fmaxf(s0, s1);
#pragma unroll
      for (int off = 1; off < 16; off <<= 1)
        rm = fmaxf(rm, __shfl_xor(rm, off, 32));
      const float mnew = fmaxf(mst[v], rm);
      const float corr = __expf(mst[v] - mnew);
      mst[v] = mnew;
      const float p0 = __expf(s0 - mnew);
      const float p1 = __expf(s1 - mnew);
      sc0[v] = p0;
      sc1[v] = p1;
      float rs = p0 + p1;
#pragma unroll
      for (int off = 1; off < 16; off <<= 1) rs += __shfl_xor(rs, off, 32);
      lst[v] = lst[v] * corr + rs;
#pragma unroll
      for (int t = 0; t < 16; ++t) acc[t][v] *= corr;
    }
    // ---- relayout P (C-layout) -> LDS -> A-fragment ----
#pragma unroll
    for (int v = 0; v < 8; ++v) {
      const int r = v + 8 * half;
      pshare[wave][r][ml] = (__bf16)sc0[v];
      pshare[wave][r][16 + ml] = (__bf16)sc1[v];
    }
    asm volatile("s_wait_dscnt 0" ::: "memory");
    const bf16x16 pa = frag_a(&pshare[wave][ml][0], 0, half);
    // ---- acc += P @ V over all 256 output dims ----
#pragma unroll
    for (int t = 0; t < 16; ++t) {
      const bf16x16 bv = frag_b(Vh + (size_t)(t * 16 + ml) * S_, kb, half);
      acc[t] = wmma_bf16(pa, bv, acc[t]);
    }
  }
  // ---- normalize + store ----
#pragma unroll
  for (int v = 0; v < 8; ++v) {
    const float inv = 1.0f / lst[v];
    const int qrow = q0 + v + 8 * half;
    __bf16* orow = O + ((size_t)bh * S_ + qrow) * 256;
#pragma unroll
    for (int t = 0; t < 16; ++t)
      orow[t * 16 + ml] = (__bf16)(acc[t][v] * inv);
  }
}

// ---------------------------------------------------------------------------
// o = o1 - lam[h]*o2 ; RMS-norm over 256 dims; * rms_scale * (1 - lam0);
// write bf16 (b, s, h*256 + d).  One wave per (b,h,s) row.
// ---------------------------------------------------------------------------
__global__ __launch_bounds__(256) void combine_rms_kernel(
    const __bf16* __restrict__ o1, const __bf16* __restrict__ o2,
    const float* __restrict__ lam, const float* __restrict__ rms_scale,
    __bf16* __restrict__ outb) {
  const int lane = threadIdx.x & 31;
  const int wave = threadIdx.x >> 5;
  const size_t r = (size_t)blockIdx.x * 8 + wave;  // bh*S + s
  const int bh = (int)(r / S_);
  const int s = (int)(r % S_);
  const int h = bh & (NH_ - 1);
  const int b = bh >> 4;
  const float lamv = lam[h];
  const float lam0 = 0.8f - 0.6f * __expf(-3.6f);

  bf16x8 a = *(const bf16x8*)(o1 + r * 256 + lane * 8);
  bf16x8 c = *(const bf16x8*)(o2 + r * 256 + lane * 8);
  float vals[8];
  float ss = 0.f;
#pragma unroll
  for (int j = 0; j < 8; ++j) {
    vals[j] = (float)a[j] - lamv * (float)c[j];
    ss += vals[j] * vals[j];
  }
#pragma unroll
  for (int off = 1; off < 32; off <<= 1) ss += __shfl_xor(ss, off, 32);
  const float rn = rsqrtf(ss * (1.0f / 256.0f) + 1e-5f);
  const float g = (1.0f - lam0) * rn;
  __bf16* dst = outb + ((size_t)(b * S_ + s)) * 4096 + h * 256 + lane * 8;
  const float* rs = rms_scale + h * 256 + lane * 8;
#pragma unroll
  for (int j = 0; j < 8; ++j) dst[j] = (__bf16)(vals[j] * g * rs[j]);
}

// ---------------------------------------------------------------------------
// Host launcher
// ---------------------------------------------------------------------------
extern "C" void kernel_launch(void* const* d_in, const int* in_sizes, int n_in,
                              void* d_out, int out_size, void* d_ws,
                              size_t ws_size, hipStream_t stream) {
  (void)in_sizes; (void)n_in; (void)out_size; (void)ws_size;
  const float* x    = (const float*)d_in[0];
  const float* w_q  = (const float*)d_in[1];
  const float* w_k  = (const float*)d_in[2];
  const float* w_v  = (const float*)d_in[3];
  const float* w_o  = (const float*)d_in[4];
  const float* lq1  = (const float*)d_in[5];
  const float* lk1  = (const float*)d_in[6];
  const float* lq2  = (const float*)d_in[7];
  const float* lk2  = (const float*)d_in[8];
  const float* rms  = (const float*)d_in[9];
  const float* fcos = (const float*)d_in[10];
  const float* fsin = (const float*)d_in[11];
  // d_in[12] = mask (causal) applied analytically in-kernel

  char* wsp = (char*)d_ws;
  auto alloc = [&](size_t bytes) -> char* {
    char* p = wsp;
    wsp += (bytes + 255) & ~(size_t)255;
    return p;
  };

  const size_t EW = (size_t)4096 * 2048;  // x / each weight elems
  const size_t EQ = (size_t)4096 * 4096;  // q/k/v elems
  const size_t EH = (size_t)2 * NH_ * S_ * HD_;  // one rope'd branch

  __bf16* x_bf  = (__bf16*)alloc(EW * 2);
  __bf16* wq_bf = (__bf16*)alloc(EW * 2);
  __bf16* wk_bf = (__bf16*)alloc(EW * 2);
  __bf16* wv_bf = (__bf16*)alloc(EW * 2);
  __bf16* wo_bf = (__bf16*)alloc(EW * 2);
  __bf16* q_bf  = (__bf16*)alloc(EQ * 2);
  __bf16* k_bf  = (__bf16*)alloc(EQ * 2);
  __bf16* v_bf  = (__bf16*)alloc(EQ * 2);
  __bf16* q1r   = (__bf16*)alloc(EH * 2);
  __bf16* q2r   = (__bf16*)alloc(EH * 2);
  __bf16* k1r   = (__bf16*)alloc(EH * 2);
  __bf16* k2r   = (__bf16*)alloc(EH * 2);
  __bf16* vT    = (__bf16*)alloc(EQ * 2);
  __bf16* o1    = (__bf16*)alloc(EQ * 2);
  __bf16* o2    = (__bf16*)alloc(EQ * 2);
  __bf16* aflat = (__bf16*)alloc(EQ * 2);
  float*  lam   = (float*)alloc(64);

  const int thr = 256;
  cvt_bf16_kernel<<<(unsigned)((EW + thr - 1) / thr), thr, 0, stream>>>(x, x_bf, EW);
  cvt_bf16_kernel<<<(unsigned)((EW + thr - 1) / thr), thr, 0, stream>>>(w_q, wq_bf, EW);
  cvt_bf16_kernel<<<(unsigned)((EW + thr - 1) / thr), thr, 0, stream>>>(w_k, wk_bf, EW);
  cvt_bf16_kernel<<<(unsigned)((EW + thr - 1) / thr), thr, 0, stream>>>(w_v, wv_bf, EW);
  cvt_bf16_kernel<<<(unsigned)((EW + thr - 1) / thr), thr, 0, stream>>>(w_o, wo_bf, EW);

  // QKV projections: (4096 x 2048) * (4096 x 2048)^T -> bf16
  dim3 gqkv(4096 / 128, 4096 / 128);
  gemm_bf16_kernel<false><<<gqkv, 256, 0, stream>>>(x_bf, wq_bf, q_bf, 4096, 4096, 2048);
  gemm_bf16_kernel<false><<<gqkv, 256, 0, stream>>>(x_bf, wk_bf, k_bf, 4096, 4096, 2048);
  gemm_bf16_kernel<false><<<gqkv, 256, 0, stream>>>(x_bf, wv_bf, v_bf, 4096, 4096, 2048);

  rope_reorg_kernel<<<(unsigned)((size_t)4096 * NH_), 256, 0, stream>>>(
      q_bf, k_bf, v_bf, fcos, fsin, q1r, q2r, k1r, k2r, vT);

  lambda_kernel<<<1, 32, 0, stream>>>(lq1, lk1, lq2, lk2, lam);

  dim3 gfa(S_ / 128, 2 * NH_);
  flash_attn_kernel<<<gfa, 256, 0, stream>>>(q1r, k1r, vT, o1);
  flash_attn_kernel<<<gfa, 256, 0, stream>>>(q2r, k2r, vT, o2);

  combine_rms_kernel<<<(unsigned)(((size_t)2 * NH_ * S_) / 8), 256, 0, stream>>>(
      o1, o2, lam, rms, aflat);

  // output projection: (4096 x 4096) * (2048 x 4096)^T -> f32 d_out
  dim3 gout(4096 / 128, 2048 / 128);
  gemm_bf16_kernel<true><<<gout, 256, 0, stream>>>(aflat, wo_bf, d_out, 4096, 2048, 4096);
}
